// QLinearMulti_83468394430856
// MI455X (gfx1250) — compile-verified, matrix operand order checked by reading, not run on hardware
//
#include <hip/hip_runtime.h>

// ---------------------------------------------------------------------------
// MI455X (gfx1250): decode multi-base quantized weight -> bf16 (L2-resident,
// 32 MB), cast x -> bf16, then wave32 WMMA GEMM (v_wmma_f32_16x16x32_bf16,
// f32 accumulation).  GEMM is compute-bound (~275 GFLOP vs ~22 us of HBM
// traffic at 23.3 TB/s), so we maximize per-wave arithmetic intensity:
// 64x64 output per wave = 4x4 grid of 16x16 tiles -> 32 FLOP/byte from L2.
// ---------------------------------------------------------------------------

typedef __attribute__((ext_vector_type(2)))  __bf16 v2bf;
typedef __attribute__((ext_vector_type(4)))  __bf16 v4bf;
typedef __attribute__((ext_vector_type(8)))  __bf16 v8bf;
typedef __attribute__((ext_vector_type(16))) __bf16 v16bf;
typedef __attribute__((ext_vector_type(8)))  float  v8f;

static constexpr int OUT_F = 4096;
static constexpr int IN_F  = 4096;
static constexpr int N_TOK = 8192;

// ---------------------------------------------------------------------------
// Kernel 1: decode quantized weight -> bf16.
// Per weight element (out, in): consecutive pair (c0,c1) from C_shadow and
// (b0,b1) from B (innermost dim = NUM_BASEWEIGHTS = 2). Each thread handles 2
// weight elements via one float4 load per tensor (16B-aligned, coalesced).
// ---------------------------------------------------------------------------
__global__ __launch_bounds__(256) void decode_weight_kernel(
    const float* __restrict__ Cs, const float* __restrict__ Bw,
    __bf16* __restrict__ Wq)
{
  const size_t e = ((size_t)blockIdx.x * blockDim.x + threadIdx.x) * 2;
  const float4 c4 = *reinterpret_cast<const float4*>(Cs + e * 2);
  const float4 b4 = *reinterpret_cast<const float4*>(Bw + e * 2);
  float w01[2];
#pragma unroll
  for (int t = 0; t < 2; ++t) {
    const float c0 = t ? c4.z : c4.x;
    const float c1 = t ? c4.w : c4.y;
    const float b0 = t ? b4.z : b4.x;
    const float b1 = t ? b4.w : b4.y;
    const float bs0 = (c0 < 0.0f) ? -b0 : b0;    // sign transfer
    const float bs1 = (c1 < 0.0f) ? -b1 : b1;
    const float R   = fabsf(c0) * bs0 + fabsf(c1) * bs1;
    const float q0  = floorf(R / bs0);           // digit 0 (already integral)
    const float rem = R - q0 * bs0;              // jnp.mod(R, bs0)
    const float q1  = rintf(rem / bs1);          // digit 1, round-half-even
    w01[t] = q0 * bs0 + q1 * bs1;                // reconstructed quant weight
  }
  v2bf o;
  o.x = (__bf16)w01[0];
  o.y = (__bf16)w01[1];
  *reinterpret_cast<v2bf*>(Wq + e) = o;
}

// ---------------------------------------------------------------------------
// Kernel 2: cast activations f32 -> bf16 (float4 load, 8B store).
// ---------------------------------------------------------------------------
__global__ __launch_bounds__(256) void cast_x_kernel(
    const float* __restrict__ X, __bf16* __restrict__ Xb)
{
  const size_t e = ((size_t)blockIdx.x * blockDim.x + threadIdx.x) * 4;
  const float4 x4 = *reinterpret_cast<const float4*>(X + e);
  v4bf o;
  o.x = (__bf16)x4.x;
  o.y = (__bf16)x4.y;
  o.z = (__bf16)x4.z;
  o.w = (__bf16)x4.w;
  *reinterpret_cast<v4bf*>(Xb + e) = o;
}

// ---------------------------------------------------------------------------
// Kernel 3: out[t,o] = sum_k Xb[t,k] * Wq[o,k] + bias[o]   (TN bf16 GEMM)
//
// Block = 256 threads = 8 wave32, wave grid 2(M) x 4(N); block tile 128x256.
// Wave tile = 64(M) x 64(N) = 4x4 subtiles of 16x16, K-step 32:
//   - CDNA5 16-bit A fragment (16x32, MxK): lane l holds row M = l&15,
//     elements 0..7 = K in [half*8, half*8+8), elements 8..15 = same + 16,
//     with half = l>>4.  Both x and W are row-major along K, so A and B
//     fragments are each two aligned global_load_b128 per lane.
//   - Accum/D layout: VGPR r, lane l -> (M = r + 8*half, N = l&15).
// Addresses are uniform-base + 32-bit per-lane offsets so the compiler can
// use saddr + scaled-voffset global loads (fewer address VGPRs).
// ---------------------------------------------------------------------------
__global__ __launch_bounds__(256) void gemm_bf16_wmma(
    const __bf16* __restrict__ Xb,   // [N_TOK][IN_F]
    const __bf16* __restrict__ Wq,   // [OUT_F][IN_F]
    const float*  __restrict__ bias, // [OUT_F]
    float*        __restrict__ out)  // [N_TOK][OUT_F]
{
  const int lane  = threadIdx.x & 31;
  const int wave  = threadIdx.x >> 5;
  const int half  = lane >> 4;   // 0 or 1
  const int r16   = lane & 15;

  const int block_m = blockIdx.y * 128;
  const int block_n = blockIdx.x * 256;
  const int wave_m  = block_m + (wave & 1) * 64;  // 2 waves along M
  const int wave_n  = block_n + (wave >> 1) * 64; // 4 waves along N

  v8f acc[4][4];
#pragma unroll
  for (int i = 0; i < 4; ++i)
#pragma unroll
    for (int j = 0; j < 4; ++j) acc[i][j] = {};

  // 32-bit element offsets from the uniform bases (max ~33.5M < 2^31).
  unsigned aoff[4];
#pragma unroll
  for (int i = 0; i < 4; ++i)
    aoff[i] = (unsigned)(wave_m + i * 16 + r16) * IN_F + half * 8;
  unsigned boff[4];
#pragma unroll
  for (int j = 0; j < 4; ++j)
    boff[j] = (unsigned)(wave_n + j * 16 + r16) * IN_F + half * 8;

#pragma unroll 2
  for (int k = 0; k < IN_F; k += 32) {
    v16bf a[4];
#pragma unroll
    for (int i = 0; i < 4; ++i) {
      const v8bf lo = *reinterpret_cast<const v8bf*>(Xb + aoff[i] + k);
      const v8bf hi = *reinterpret_cast<const v8bf*>(Xb + aoff[i] + k + 16);
      a[i] = __builtin_shufflevector(lo, hi, 0, 1, 2, 3, 4, 5, 6, 7,
                                     8, 9, 10, 11, 12, 13, 14, 15);
      // Stream the activations: pull the tile 8 K-iterations ahead into L2.
      __builtin_prefetch(Xb + aoff[i] + k + 256, 0, 1);
    }
    v16bf b[4];
#pragma unroll
    for (int j = 0; j < 4; ++j) {
      const v8bf lo = *reinterpret_cast<const v8bf*>(Wq + boff[j] + k);
      const v8bf hi = *reinterpret_cast<const v8bf*>(Wq + boff[j] + k + 16);
      b[j] = __builtin_shufflevector(lo, hi, 0, 1, 2, 3, 4, 5, 6, 7,
                                     8, 9, 10, 11, 12, 13, 14, 15);
    }
#pragma unroll
    for (int i = 0; i < 4; ++i)
#pragma unroll
      for (int j = 0; j < 4; ++j)
        acc[i][j] = __builtin_amdgcn_wmma_f32_16x16x32_bf16(
            false, a[i], false, b[j], (short)0, acc[i][j], false, false);
  }

  // Epilogue: bias add + f32 store. Lanes 0..15 cover 64 contiguous bytes
  // per (i, r) row -> coalesced half-wave stores.
#pragma unroll
  for (int j = 0; j < 4; ++j) {
    const int col = wave_n + j * 16 + r16;
    const float bv = bias[col];
#pragma unroll
    for (int i = 0; i < 4; ++i) {
#pragma unroll
      for (int r = 0; r < 8; ++r) {
        const int row = wave_m + i * 16 + half * 8 + r;
        out[(size_t)row * OUT_F + col] = acc[i][j][r] + bv;
      }
    }
  }
}

// ---------------------------------------------------------------------------
// Launch: inputs are (x, C_shadow, B, bias) in setup_inputs() order.
// Workspace: [0, 32MB) = Wq bf16, [32MB, 96MB) = Xb bf16.
// ---------------------------------------------------------------------------
extern "C" void kernel_launch(void* const* d_in, const int* in_sizes, int n_in,
                              void* d_out, int out_size, void* d_ws, size_t ws_size,
                              hipStream_t stream) {
  const float* x    = (const float*)d_in[0];
  const float* Cs   = (const float*)d_in[1];
  const float* Bw   = (const float*)d_in[2];
  const float* bias = (const float*)d_in[3];
  float* out = (float*)d_out;

  __bf16* Wq = (__bf16*)d_ws;
  __bf16* Xb = (__bf16*)((char*)d_ws + (size_t)OUT_F * IN_F * sizeof(__bf16));

  // 1) Decode weight -> bf16. 16M elements, 2 per thread.
  {
    const int nthreads = (OUT_F * IN_F) / 2;
    decode_weight_kernel<<<nthreads / 256, 256, 0, stream>>>(Cs, Bw, Wq);
  }
  // 2) Cast x -> bf16. 33.5M elements, 4 per thread.
  {
    const int nthreads = (N_TOK * IN_F) / 4;
    cast_x_kernel<<<nthreads / 256, 256, 0, stream>>>(x, Xb);
  }
  // 3) WMMA GEMM: grid (N/256, M/128) = (16, 64), 256 threads/block.
  {
    dim3 grid(OUT_F / 256, N_TOK / 128);
    gemm_bf16_wmma<<<grid, 256, 0, stream>>>(Xb, Wq, bias, out);
  }
}
